// AttentionHead_20650202759186
// MI455X (gfx1250) — matrix operand order repaired and checked by reading, not compiled, not run on hardware
//
#include <hip/hip_runtime.h>
#include <hip/hip_bf16.h>

// ---------------------------------------------------------------------------
// Causal single-head attention, flash-attention style, for MI455X (gfx1250).
// B=4, T=4096, C=1024, H=64. fp32 in/out, f16 WMMA (f32 accum) internally.
// ---------------------------------------------------------------------------

#define BATCH 4
#define SEQ   4096
#define NEMB  1024
#define HEAD  64

typedef _Float16 h8  __attribute__((ext_vector_type(8)));
typedef _Float16 h16 __attribute__((ext_vector_type(16)));
typedef float    f8  __attribute__((ext_vector_type(8)));
typedef float    f4v __attribute__((ext_vector_type(4)));

static __device__ __forceinline__ f8 wmma_f16(h16 a, h16 b, f8 c) {
    return __builtin_amdgcn_wmma_f32_16x16x32_f16(
        /*neg_a=*/false, a, /*neg_b=*/false, b,
        /*c_mod=*/(short)0, c, /*reuse_a=*/false, /*reuse_b=*/false);
}

static __device__ __forceinline__ h16 cat8(h8 lo, h8 hi) {
    return __builtin_shufflevector(lo, hi, 0, 1, 2, 3, 4, 5, 6, 7,
                                           8, 9, 10, 11, 12, 13, 14, 15);
}

// --------------------------------------------------------------------------
// Kernel 0: W[C][H] f32  ->  Wt[H][C] f16  (x3 matrices)
// --------------------------------------------------------------------------
__global__ __launch_bounds__(256) void ah_prep_weights(
    const float* __restrict__ Wk, const float* __restrict__ Wq,
    const float* __restrict__ Wv,
    _Float16* __restrict__ Wkt, _Float16* __restrict__ Wqt,
    _Float16* __restrict__ Wvt) {
    int e = blockIdx.x * 256 + threadIdx.x;      // 65536 elements per matrix
    if (e >= NEMB * HEAD) return;
    int h = e & (HEAD - 1);
    int c = e >> 6;
    Wkt[h * NEMB + c] = (_Float16)Wk[c * HEAD + h];
    Wqt[h * NEMB + c] = (_Float16)Wq[c * HEAD + h];
    Wvt[h * NEMB + c] = (_Float16)Wv[c * HEAD + h];
}

// --------------------------------------------------------------------------
// Kernel 1: QKV projection.  One wave = 16 token rows x all 64 head cols,
// for K, Q, V simultaneously (x A-fragment reused 12x).
//   Kf[b*T+t][h] f16, Qf[b*T+t][h] f16 (pre-scaled by 1/8), Vt[b][h][t] f16.
//
// Weights are staged per-block in LDS, double-buffered: 12 KB/k-chunk loaded
// cooperatively once (vs 8x redundantly per wave), so the WMMA inner loop
// depends only on short-latency ds_loads. fp32 A-chunk loads for the next
// iteration issue before the current WMMAs (HBM latency overlap).
// --------------------------------------------------------------------------
#define WROW 40                      // padded halves per weight row (80 B)
#define WMAT (64 * WROW)             // halves per matrix per buffer

__global__ __launch_bounds__(256) void ah_proj(
    const float* __restrict__ x,
    const _Float16* __restrict__ Wkt, const _Float16* __restrict__ Wqt,
    const _Float16* __restrict__ Wvt,
    const float* __restrict__ bk, const float* __restrict__ bq,
    const float* __restrict__ bv,
    _Float16* __restrict__ Kf, _Float16* __restrict__ Qf,
    _Float16* __restrict__ Vt) {
    __shared__ __align__(16) _Float16 wlds[2][3 * WMAT];   // 2 x 15 KB

    const int tid  = threadIdx.x;
    const int wave = tid >> 5;
    const int lane = tid & 31;
    const int l16  = lane & 15;
    const int hi   = lane >> 4;

    const int tile = blockIdx.x * 8 + wave;        // 1024 tiles of 16 rows
    const int R0   = tile * 16;

    // ---- cooperative weight-fill mapping: 256 threads x (3 x 16 B) ----
    const int    fn   = tid >> 2;                  // n-row 0..63
    const int    fkc  = (tid & 3) * 8;             // k sub-chunk 0,8,16,24
    const size_t fsrc = (size_t)fn * NEMB + fkc;   // + k0 at use site
    const int    fdst = fn * WROW + fkc;

    f8 accK[4], accQ[4], accV[4];
#pragma unroll
    for (int nt = 0; nt < 4; ++nt) { accK[nt] = (f8)0.f; accQ[nt] = (f8)0.f; accV[nt] = (f8)0.f; }

    const float* xrow = x + (size_t)(R0 + l16) * NEMB;
    const int aoff = hi ? 16 : 0;   // A-frag K sub-offset per half-wave
    const int boff = hi ? 16 : 0;   // B-frag K sub-offset per half-wave

    // ---- prologue: fill weight buffer 0 (k0 = 0) ----
    {
        h8 f0 = *(const h8*)(Wkt + fsrc);
        h8 f1 = *(const h8*)(Wqt + fsrc);
        h8 f2 = *(const h8*)(Wvt + fsrc);
        *(h8*)(&wlds[0][0 * WMAT + fdst]) = f0;
        *(h8*)(&wlds[0][1 * WMAT + fdst]) = f1;
        *(h8*)(&wlds[0][2 * WMAT + fdst]) = f2;
    }

    // ---- prologue: raw fp32 A-chunk for k0 = 0 ----
    f4v xr0, xr1, xr2, xr3;
    {
        const f4v* p1 = (const f4v*)(xrow + aoff);
        const f4v* p2 = (const f4v*)(xrow + aoff + 8);
        xr0 = p1[0]; xr1 = p1[1]; xr2 = p2[0]; xr3 = p2[1];
    }
    __syncthreads();

    for (int k0 = 0; k0 < NEMB; k0 += 32) {
        const int  cur  = (k0 >> 5) & 1;
        const bool more = (k0 + 32) < NEMB;

        // ---- issue next weight-chunk global loads early (into registers) ----
        h8 f0, f1, f2;
        if (more) {
            f0 = *(const h8*)(Wkt + fsrc + k0 + 32);
            f1 = *(const h8*)(Wqt + fsrc + k0 + 32);
            f2 = *(const h8*)(Wvt + fsrc + k0 + 32);
        }
        // ---- issue next A-chunk fp32 loads early ----
        const int kn = (k0 + 32) & (NEMB - 1);        // wraps to 0 on last iter
        const f4v* np1 = (const f4v*)(xrow + kn + aoff);
        const f4v* np2 = (const f4v*)(xrow + kn + aoff + 8);
        f4v nx0 = np1[0], nx1 = np1[1], nx2 = np2[0], nx3 = np2[1];

        // ---- current A fragment (16x32 f16) ----
        h16 a;
#pragma unroll
        for (int i = 0; i < 4; ++i) {
            a[i]      = (_Float16)xr0[i];
            a[4 + i]  = (_Float16)xr1[i];
            a[8 + i]  = (_Float16)xr2[i];
            a[12 + i] = (_Float16)xr3[i];
        }

        // ---- 12 WMMAs; B fragments from LDS (short-latency ds_loads) ----
        const _Float16* wb = &wlds[cur][0];
        const int frow = l16 * WROW + boff;          // within the nt/matrix region
#pragma unroll
        for (int nt = 0; nt < 4; ++nt) {
            h16 bK = *(const h16*)(wb + 0 * WMAT + nt * 16 * WROW + frow);
            accK[nt] = wmma_f16(a, bK, accK[nt]);
        }
#pragma unroll
        for (int nt = 0; nt < 4; ++nt) {
            h16 bQ = *(const h16*)(wb + 1 * WMAT + nt * 16 * WROW + frow);
            accQ[nt] = wmma_f16(a, bQ, accQ[nt]);
        }
#pragma unroll
        for (int nt = 0; nt < 4; ++nt) {
            h16 bV = *(const h16*)(wb + 2 * WMAT + nt * 16 * WROW + frow);
            accV[nt] = wmma_f16(a, bV, accV[nt]);
        }

        // ---- commit next weight chunk to the other buffer, then sync ----
        if (more) {
            _Float16* wn = &wlds[cur ^ 1][0];
            *(h8*)(wn + 0 * WMAT + fdst) = f0;
            *(h8*)(wn + 1 * WMAT + fdst) = f1;
            *(h8*)(wn + 2 * WMAT + fdst) = f2;
        }
        xr0 = nx0; xr1 = nx1; xr2 = nx2; xr3 = nx3;
        __syncthreads();
    }

    // ---- bias add, scale Q, stores (C-layout: VGPR r <-> row r+8*hi) ----
#pragma unroll
    for (int nt = 0; nt < 4; ++nt) {
        const int n = nt * 16 + l16;
        const float vbk = bk[n], vbq = bq[n], vbv = bv[n];
#pragma unroll
        for (int r = 0; r < 8; ++r) {
            const int row = R0 + r + hi * 8;              // flat b*T+t
            Kf[(size_t)row * HEAD + n] = (_Float16)(accK[nt][r] + vbk);
            Qf[(size_t)row * HEAD + n] = (_Float16)((accQ[nt][r] + vbq) * 0.125f);
            const int b = row >> 12;                      // /SEQ
            const int t = row & (SEQ - 1);
            Vt[((size_t)b * HEAD + n) * SEQ + t] = (_Float16)(accV[nt][r] + vbv);
        }
    }
}

// --------------------------------------------------------------------------
// Kernel 2: causal flash attention. One wave = one 16-query tile.
// --------------------------------------------------------------------------
#define PSTRIDE 40   // halves per P row in LDS (80B, avoids bank hotspots, 16B aligned)

static __device__ __forceinline__ float rmax16(float v) {
#pragma unroll
    for (int m = 1; m <= 8; m <<= 1) v = fmaxf(v, __shfl_xor(v, m, 32));
    return v;
}
static __device__ __forceinline__ float rsum16(float v) {
#pragma unroll
    for (int m = 1; m <= 8; m <<= 1) v += __shfl_xor(v, m, 32);
    return v;
}

__global__ __launch_bounds__(256) void ah_attn(
    const _Float16* __restrict__ Qf, const _Float16* __restrict__ Kf,
    const _Float16* __restrict__ Vt, float* __restrict__ out) {
    __shared__ __align__(16) _Float16 Plds[8][16 * PSTRIDE];

    const int wave = threadIdx.x >> 5;
    const int lane = threadIdx.x & 31;
    const int l16  = lane & 15;
    const int hi   = lane >> 4;
    const int aoff = hi ? 16 : 0;
    const int boff = hi ? 16 : 0;

    const int tile = blockIdx.x * 8 + wave;      // 1024 tiles (256 per batch)
    const int b    = tile >> 8;
    const int q0   = (tile & 255) * 16;

    // ---- load the two Q A-fragments (16x32 each), reused for every chunk ----
    const _Float16* qrow = Qf + ((size_t)b * SEQ + q0 + l16) * HEAD;
    h16 qa[2];
#pragma unroll
    for (int c = 0; c < 2; ++c) {
        h8 lo = *(const h8*)(qrow + c * 32 + aoff);
        h8 hv = *(const h8*)(qrow + c * 32 + aoff + 8);
        qa[c] = cat8(lo, hv);
    }

    f8 O[4];
#pragma unroll
    for (int nt = 0; nt < 4; ++nt) O[nt] = (f8)0.f;
    float mrow[8], lrow[8];
#pragma unroll
    for (int r = 0; r < 8; ++r) { mrow[r] = -1e30f; lrow[r] = 0.f; }

    const _Float16* Kb = Kf + (size_t)b * SEQ * HEAD;
    const _Float16* Vb = Vt + (size_t)b * HEAD * SEQ;
    _Float16* pl = &Plds[wave][0];

    for (int k0 = 0; k0 <= q0; k0 += 32) {       // keys [k0, k0+32)
        __builtin_prefetch(Kb + (size_t)(k0 + 32 + l16) * HEAD, 0, 1);
        __builtin_prefetch(Vb + (size_t)l16 * SEQ + k0 + 32, 0, 1);

        // ---- scores: S = (Q/8) . K^T, two 16-key tiles, K-dim = HEAD = 64 ----
        f8 S0 = (f8)0.f, S1 = (f8)0.f;
        {
            const _Float16* kr0 = Kb + (size_t)(k0 + l16) * HEAD;
            const _Float16* kr1 = Kb + (size_t)(k0 + 16 + l16) * HEAD;
            h16 kb00 = *(const h16*)(kr0 + boff);
            h16 kb01 = *(const h16*)(kr0 + 32 + boff);
            h16 kb10 = *(const h16*)(kr1 + boff);
            h16 kb11 = *(const h16*)(kr1 + 32 + boff);
            S0 = wmma_f16(qa[0], kb00, S0);
            S0 = wmma_f16(qa[1], kb01, S0);
            S1 = wmma_f16(qa[0], kb10, S1);
            S1 = wmma_f16(qa[1], kb11, S1);
        }

        // ---- causal mask + online softmax (row = r + 8*hi across 16 lanes) ----
        const int kA = k0 + l16;
        const int kB = k0 + 16 + l16;
        float alpha[8];
#pragma unroll
        for (int r = 0; r < 8; ++r) {
            const int q = q0 + r + hi * 8;
            float s0 = (kA > q) ? -1e30f : S0[r];
            float s1 = (kB > q) ? -1e30f : S1[r];
            float rm = rmax16(fmaxf(s0, s1));
            float mn = fmaxf(mrow[r], rm);
            alpha[r] = __expf(mrow[r] - mn);
            float p0 = __expf(s0 - mn);
            float p1 = __expf(s1 - mn);
            S0[r] = p0;
            S1[r] = p1;
            lrow[r] = lrow[r] * alpha[r] + rsum16(p0 + p1);
            mrow[r] = mn;
        }
#pragma unroll
        for (int nt = 0; nt < 4; ++nt)
#pragma unroll
            for (int r = 0; r < 8; ++r) O[nt][r] *= alpha[r];

        // ---- stage P (16x32 f16) through LDS: C-layout -> A-fragment ----
#pragma unroll
        for (int r = 0; r < 8; ++r) {
            const int m = r + hi * 8;
            pl[m * PSTRIDE + l16]      = (_Float16)S0[r];
            pl[m * PSTRIDE + 16 + l16] = (_Float16)S1[r];
        }
        asm volatile("s_wait_dscnt 0" ::: "memory");   // same-wave LDS RAW fence
        h8 plo = *(const h8*)(pl + l16 * PSTRIDE + aoff);
        h8 phi = *(const h8*)(pl + l16 * PSTRIDE + aoff + 8);
        asm volatile("" ::: "memory");                 // pin load-before-next-store
        h16 pa = cat8(plo, phi);

        // ---- O += P . V : batch the four V B-fragments, then four WMMAs ----
        {
            h16 vb0 = *(const h16*)(Vb + ((size_t)(l16)) * SEQ + k0 + boff);
            h16 vb1 = *(const h16*)(Vb + ((size_t)(16 + l16)) * SEQ + k0 + boff);
            h16 vb2 = *(const h16*)(Vb + ((size_t)(32 + l16)) * SEQ + k0 + boff);
            h16 vb3 = *(const h16*)(Vb + ((size_t)(48 + l16)) * SEQ + k0 + boff);
            O[0] = wmma_f16(pa, vb0, O[0]);
            O[1] = wmma_f16(pa, vb1, O[1]);
            O[2] = wmma_f16(pa, vb2, O[2]);
            O[3] = wmma_f16(pa, vb3, O[3]);
        }
    }

    // ---- normalize and store fp32 output ----
    float* orow = out + ((size_t)b * SEQ + q0) * HEAD;
#pragma unroll
    for (int r = 0; r < 8; ++r) {
        const float inv = 1.0f / lrow[r];
        const int m = r + hi * 8;
#pragma unroll
        for (int nt = 0; nt < 4; ++nt) {
            const int n = nt * 16 + l16;
            orow[(size_t)m * HEAD + n] = O[nt][r] * inv;
        }
    }
}

// --------------------------------------------------------------------------
extern "C" void kernel_launch(void* const* d_in, const int* in_sizes, int n_in,
                              void* d_out, int out_size, void* d_ws, size_t ws_size,
                              hipStream_t stream) {
    const float* x  = (const float*)d_in[0];
    const float* Wk = (const float*)d_in[1];
    const float* bk = (const float*)d_in[2];
    const float* Wq = (const float*)d_in[3];
    const float* bq = (const float*)d_in[4];
    const float* Wv = (const float*)d_in[5];
    const float* bv = (const float*)d_in[6];
    float* out = (float*)d_out;

    char* ws = (char*)d_ws;
    const size_t WT_BYTES  = (size_t)HEAD * NEMB * sizeof(_Float16);        // 128 KB
    const size_t KQV_BYTES = (size_t)BATCH * SEQ * HEAD * sizeof(_Float16); // 2 MB
    _Float16* Wkt = (_Float16*)(ws);
    _Float16* Wqt = (_Float16*)(ws + WT_BYTES);
    _Float16* Wvt = (_Float16*)(ws + 2 * WT_BYTES);
    _Float16* Kf  = (_Float16*)(ws + 3 * WT_BYTES);
    _Float16* Qf  = (_Float16*)(ws + 3 * WT_BYTES + KQV_BYTES);
    _Float16* Vt  = (_Float16*)(ws + 3 * WT_BYTES + 2 * KQV_BYTES);

    ah_prep_weights<<<(NEMB * HEAD + 255) / 256, 256, 0, stream>>>(
        Wk, Wq, Wv, Wkt, Wqt, Wvt);

    // 1024 row tiles, 8 waves (256 threads) per block
    ah_proj<<<128, 256, 0, stream>>>(x, Wkt, Wqt, Wvt, bk, bq, bv, Kf, Qf, Vt);

    // 1024 query tiles, 8 waves per block
    ah_attn<<<128, 256, 0, stream>>>(Qf, Kf, Vt, out);
}
